// SpatialAttention_24713241821723
// MI455X (gfx1250) — matrix operand order repaired and checked by reading, not compiled, not run on hardware
//
#include <hip/hip_runtime.h>
#include <hip/hip_bf16.h>

typedef __attribute__((ext_vector_type(16))) _Float16 v16h;
typedef __attribute__((ext_vector_type(8)))  _Float16 v8h;
typedef __attribute__((ext_vector_type(8)))  float    v8f;

#define WMMA_F32_F16(A,B,C) \
  __builtin_amdgcn_wmma_f32_16x16x32_f16(false,(A),false,(B),(short)0,(C),false,false)

// Problem constants (B=2, H=W=64, C=256, 8 heads, hd=32)
constexpr int Bv = 2, Nv = 4096, Cv = 256, NHv = 8, HDv = 32;
constexpr int TOK = Bv * Nv;                 // 8192 tokens
// softmax scale folded into Q at projection time, in base-2 domain:
// (1/sqrt(32)) * log2(e)
constexpr float QSCALE = 0.17677669529663688f * 1.44269504088896340736f;

// ---------------------------------------------------------------------------
// Stage 1: fp32 -> fp16 conversion / weight transposes
// ---------------------------------------------------------------------------
__global__ void cvt_x_kernel(const float* __restrict__ x, _Float16* __restrict__ xh) {
  int i = blockIdx.x * 256 + threadIdx.x;     // covers TOK*Cv = 2,097,152
  xh[i] = (_Float16)x[i];
}

__global__ void cvt_wqkv_kernel(const float* __restrict__ w, _Float16* __restrict__ wt) {
  int i = blockIdx.x * 256 + threadIdx.x;     // covers 768*256 = 196,608
  int c = i & 255;                            // input-channel
  int d = i >> 8;                             // output col in [0,768)
  wt[d * 256 + c] = (_Float16)w[c * 768 + d]; // wt: [768, 256]
}

__global__ void cvt_wproj_kernel(const float* __restrict__ w, _Float16* __restrict__ wt) {
  int i = blockIdx.x * 256 + threadIdx.x;     // covers 256*256 = 65,536
  int c = i & 255;
  int d = i >> 8;
  wt[d * 256 + c] = (_Float16)w[c * 256 + d]; // wt: [256, 256]
}

// ---------------------------------------------------------------------------
// Stage 2: fused QKV GEMM  [8192,256] x [256,768] -> scatter Q,K,[V^T] (f16)
// one wave per 16x16 output tile; K looped in 32-chunks (8 wmma / wave).
// Q section is pre-scaled by QSCALE so the attention loop needs no score scaling.
// ---------------------------------------------------------------------------
__global__ __launch_bounds__(128) void qkv_gemm_kernel(
    const _Float16* __restrict__ xh, const _Float16* __restrict__ wqkvT,
    const float* __restrict__ bqkv,
    _Float16* __restrict__ qb, _Float16* __restrict__ kb2, _Float16* __restrict__ vtb) {
  int wave = blockIdx.x * 4 + (threadIdx.x >> 5);   // [0, 24576)
  int lane = threadIdx.x & 31;
  int g = lane >> 4, n = lane & 15;
  int rt = wave / 48;                               // row tile  [0,512)
  int ct = wave % 48;                               // col tile  [0,48)
  const _Float16* arow = xh    + (size_t)(rt * 16 + n) * 256;
  const _Float16* brow = wqkvT + (size_t)(ct * 16 + n) * 256;
  v8f acc = {};
#pragma unroll
  for (int kk = 0; kk < 256; kk += 32) {
    v8h alo = *(const v8h*)(arow + kk + g * 16);       // A: K = g*16 + 0..7
    v8h ahi = *(const v8h*)(arow + kk + g * 16 + 8);   // A: K = g*16 + 8..15
    v16h a;
#pragma unroll
    for (int i = 0; i < 8; i++) { a[i] = alo[i]; a[8 + i] = ahi[i]; }
    v16h b = *(const v16h*)(brow + kk + g * 16);       // B: K = g*16 + 0..15
    acc = WMMA_F32_F16(a, b, acc);
  }
  int col = ct * 16 + n;                 // output column [0,768)
  float bias = bqkv[col];
  int sect = col >> 8;                   // 0=Q 1=K 2=V (uniform per wave tile)
  float vscale = (sect == 0) ? QSCALE : 1.0f;
  int cc = col & 255;
  int h = cc >> 5, hd = cc & 31;
#pragma unroll
  for (int r = 0; r < 8; r++) {
    int row = rt * 16 + 8 * g + r;       // token row [0,8192)
    int bb = row >> 12, tok = row & 4095;
    int bh = bb * NHv + h;
    _Float16 val = (_Float16)((acc[r] + bias) * vscale);
    if (sect == 0)      qb [(size_t)bh * Nv * HDv + (size_t)tok * HDv + hd] = val;
    else if (sect == 1) kb2[(size_t)bh * Nv * HDv + (size_t)tok * HDv + hd] = val;
    else                vtb[(size_t)bh * HDv * Nv + (size_t)hd * Nv + tok]  = val; // V^T
  }
}

// ---------------------------------------------------------------------------
// Stage 3: flash attention. One wave per (bh, 16-query tile), 64 keys/iter.
// S^T = K(A) x Q^T(B): the S^T D-layout (lane = query col, regs = keys) is
// bit-identical to the 16x32 P A-fragment layout, so PV needs no transpose.
// rel_bias is a per-head scalar -> cancels under softmax shift-invariance.
// ---------------------------------------------------------------------------
__global__ __launch_bounds__(128) void flash_attn_kernel(
    const _Float16* __restrict__ qb, const _Float16* __restrict__ kb2,
    const _Float16* __restrict__ vtb, _Float16* __restrict__ oh) {
  int wave = blockIdx.x * 4 + (threadIdx.x >> 5);   // [0, 4096)
  int lane = threadIdx.x & 31;
  int g = lane >> 4, n = lane & 15;
  int bh = wave >> 8;                               // [0,16)
  int qbase = (wave & 255) * 16;
  const _Float16* qp = qb  + (size_t)bh * Nv * HDv;
  const _Float16* kp = kb2 + (size_t)bh * Nv * HDv;
  const _Float16* vp = vtb + (size_t)bh * HDv * Nv;

  // Q^T B-fragment: col n = query qbase+n, K(=hd) = g*16..g*16+15 contiguous
  v16h qf = *(const v16h*)(qp + (size_t)(qbase + n) * HDv + g * 16);

  v8f o0 = {}, o1 = {};
  float m = -3.0e38f, l = 0.0f;

  for (int kk = 0; kk < Nv; kk += 64) {
    // K A-fragments for four 16-key sub-tiles (keys kk+16t .. kk+16t+15)
    v16h a0, a1, a2, a3;
    {
      const _Float16* kr0 = kp + (size_t)(kk + n) * HDv;
      const _Float16* kr1 = kp + (size_t)(kk + 16 + n) * HDv;
      const _Float16* kr2 = kp + (size_t)(kk + 32 + n) * HDv;
      const _Float16* kr3 = kp + (size_t)(kk + 48 + n) * HDv;
      v8h l0 = *(const v8h*)(kr0 + g * 16), h0 = *(const v8h*)(kr0 + g * 16 + 8);
      v8h l1 = *(const v8h*)(kr1 + g * 16), h1 = *(const v8h*)(kr1 + g * 16 + 8);
      v8h l2 = *(const v8h*)(kr2 + g * 16), h2 = *(const v8h*)(kr2 + g * 16 + 8);
      v8h l3 = *(const v8h*)(kr3 + g * 16), h3 = *(const v8h*)(kr3 + g * 16 + 8);
#pragma unroll
      for (int i = 0; i < 8; i++) {
        a0[i] = l0[i]; a0[8 + i] = h0[i];
        a1[i] = l1[i]; a1[8 + i] = h1[i];
        a2[i] = l2[i]; a2[8 + i] = h2[i];
        a3[i] = l3[i]; a3[8 + i] = h3[i];
      }
    }
    if (kk + 64 < Nv) {   // uniform branch; speculative prefetch of next block
      __builtin_prefetch(kp + (size_t)(kk + 64 + n) * HDv, 0, 0);
      __builtin_prefetch(vp + (size_t)n * Nv + kk + 64, 0, 0);
    }
    v8f zc = {};
    v8f s0 = WMMA_F32_F16(a0, qf, zc);   // S^T tile t: reg r = score(query n, key kk+16t+8g+r)
    v8f s1 = WMMA_F32_F16(a1, qf, zc);   // (already in base-2 log domain via QSCALE)
    v8f s2 = WMMA_F32_F16(a2, qf, zc);
    v8f s3 = WMMA_F32_F16(a3, qf, zc);

    // per-query (lane col n) online softmax
    float bm = -3.0e38f;
#pragma unroll
    for (int r = 0; r < 8; r++)
      bm = fmaxf(bm, fmaxf(fmaxf(s0[r], s1[r]), fmaxf(s2[r], s3[r])));
    bm = fmaxf(bm, __shfl_xor(bm, 16, 32));
    float mn = fmaxf(m, bm);
    float alpha = exp2f(m - mn);
    float ls = 0.0f;
    v16h pa01, pa23;                      // P as two 16x32 A-fragments (direct!)
#pragma unroll
    for (int r = 0; r < 8; r++) {
      float p0 = exp2f(s0[r] - mn);
      float p1 = exp2f(s1[r] - mn);
      float p2 = exp2f(s2[r] - mn);
      float p3 = exp2f(s3[r] - mn);
      ls += (p0 + p1) + (p2 + p3);
      pa01[r] = (_Float16)p0; pa01[8 + r] = (_Float16)p1;
      pa23[r] = (_Float16)p2; pa23[8 + r] = (_Float16)p3;
    }
    ls += __shfl_xor(ls, 16, 32);
    l = l * alpha + ls;
    m = mn;
    // rescale O: row of O-tile is query 8g+r -> fetch that query's alpha
#pragma unroll
    for (int r = 0; r < 8; r++) {
      float ar = __shfl(alpha, (g << 3) | r, 32);
      o0[r] *= ar; o1[r] *= ar;
    }
    // V B-fragments from V^T (contiguous): col = hd, K = key within 32-block
    v16h vb00 = __builtin_shufflevector(*(const v8h*)(vp + (size_t)n * Nv + kk + g * 8), *(const v8h*)(vp + (size_t)n * Nv + kk + 16 + g * 8), 0,1,2,3,4,5,6,7,8,9,10,11,12,13,14,15);             // hd 0-15,  keys kk..kk+31
    v16h vb10 = __builtin_shufflevector(*(const v8h*)(vp + (size_t)(16 + n) * Nv + kk + g * 8), *(const v8h*)(vp + (size_t)(16 + n) * Nv + kk + 16 + g * 8), 0,1,2,3,4,5,6,7,8,9,10,11,12,13,14,15);      // hd 16-31, keys kk..kk+31
    v16h vb01 = __builtin_shufflevector(*(const v8h*)(vp + (size_t)n * Nv + kk + 32 + g * 8), *(const v8h*)(vp + (size_t)n * Nv + kk + 48 + g * 8), 0,1,2,3,4,5,6,7,8,9,10,11,12,13,14,15);        // hd 0-15,  keys kk+32..kk+63
    v16h vb11 = __builtin_shufflevector(*(const v8h*)(vp + (size_t)(16 + n) * Nv + kk + 32 + g * 8), *(const v8h*)(vp + (size_t)(16 + n) * Nv + kk + 48 + g * 8), 0,1,2,3,4,5,6,7,8,9,10,11,12,13,14,15); // hd 16-31, keys kk+32..kk+63
    o0 = WMMA_F32_F16(pa01, vb00, o0);
    o1 = WMMA_F32_F16(pa01, vb10, o1);
    o0 = WMMA_F32_F16(pa23, vb01, o0);
    o1 = WMMA_F32_F16(pa23, vb11, o1);
  }

  float inv = 1.0f / l;
  int b = bh >> 3, h = bh & 7;
#pragma unroll
  for (int r = 0; r < 8; r++) {
    float ir = __shfl(inv, (g << 3) | r, 32);
    size_t row = (size_t)(b * Nv + qbase + 8 * g + r);
    oh[row * 256 + h * 32 + n]      = (_Float16)(o0[r] * ir);
    oh[row * 256 + h * 32 + 16 + n] = (_Float16)(o1[r] * ir);
  }
}

// ---------------------------------------------------------------------------
// Stage 4: output projection [8192,256] x [256,256] + bias -> fp32
// ---------------------------------------------------------------------------
__global__ __launch_bounds__(128) void proj_gemm_kernel(
    const _Float16* __restrict__ oh, const _Float16* __restrict__ wprojT,
    const float* __restrict__ bproj, float* __restrict__ out) {
  int wave = blockIdx.x * 4 + (threadIdx.x >> 5);   // [0, 8192)
  int lane = threadIdx.x & 31;
  int g = lane >> 4, n = lane & 15;
  int rt = wave >> 4;                               // [0,512)
  int ct = wave & 15;                               // [0,16)
  const _Float16* arow = oh     + (size_t)(rt * 16 + n) * 256;
  const _Float16* brow = wprojT + (size_t)(ct * 16 + n) * 256;
  v8f acc = {};
#pragma unroll
  for (int kk = 0; kk < 256; kk += 32) {
    v8h alo = *(const v8h*)(arow + kk + g * 16);
    v8h ahi = *(const v8h*)(arow + kk + g * 16 + 8);
    v16h a;
#pragma unroll
    for (int i = 0; i < 8; i++) { a[i] = alo[i]; a[8 + i] = ahi[i]; }
    v16h b = *(const v16h*)(brow + kk + g * 16);
    acc = WMMA_F32_F16(a, b, acc);
  }
  int col = ct * 16 + n;
  float bias = bproj[col];
#pragma unroll
  for (int r = 0; r < 8; r++) {
    int row = rt * 16 + 8 * g + r;
    out[(size_t)row * 256 + col] = acc[r] + bias;
  }
}

// ---------------------------------------------------------------------------
extern "C" void kernel_launch(void* const* d_in, const int* in_sizes, int n_in,
                              void* d_out, int out_size, void* d_ws, size_t ws_size,
                              hipStream_t stream) {
  (void)in_sizes; (void)n_in; (void)out_size; (void)ws_size;
  const float* x      = (const float*)d_in[0];
  const float* w_qkv  = (const float*)d_in[1];
  const float* b_qkv  = (const float*)d_in[2];
  const float* w_proj = (const float*)d_in[3];
  const float* b_proj = (const float*)d_in[4];
  // d_in[5] = rel_bias: per-head scalar added pre-softmax -> softmax shift
  // invariance means it cannot affect the output; intentionally unused.
  float* out = (float*)d_out;

  char* ws = (char*)d_ws;
  _Float16* xh     = (_Float16*)(ws + 0);          // 8192*256*2  = 4,194,304
  _Float16* wqkvT  = (_Float16*)(ws + 4194304);    // 768*256*2   =   393,216
  _Float16* wprojT = (_Float16*)(ws + 4587520);    // 256*256*2   =   131,072
  _Float16* qb     = (_Float16*)(ws + 4718592);    // 16*4096*32*2= 4,194,304
  _Float16* kb2    = (_Float16*)(ws + 8912896);    //             = 4,194,304
  _Float16* vtb    = (_Float16*)(ws + 13107200);   // V^T         = 4,194,304
  _Float16* oh     = (_Float16*)(ws + 17301504);   // 8192*256*2  = 4,194,304
                                                   // total ~20.5 MB

  cvt_x_kernel    <<<TOK * Cv / 256, 256, 0, stream>>>(x, xh);
  cvt_wqkv_kernel <<<768,            256, 0, stream>>>(w_qkv, wqkvT);
  cvt_wproj_kernel<<<256,            256, 0, stream>>>(w_proj, wprojT);

  // 512 row-tiles * 48 col-tiles = 24576 waves, 4 waves/block
  qkv_gemm_kernel <<<6144, 128, 0, stream>>>(xh, wqkvT, b_qkv, qb, kb2, vtb);

  // 16 (b,h) * 256 query-tiles = 4096 waves
  flash_attn_kernel<<<1024, 128, 0, stream>>>(qb, kb2, vtb, oh);

  // 512 row-tiles * 16 col-tiles = 8192 waves
  proj_gemm_kernel<<<2048, 128, 0, stream>>>(oh, wprojT, b_proj, out);
}
